// MyTripletLoss_63144609185907
// MI455X (gfx1250) — matrix operand order repaired and checked by reading, not compiled, not run on hardware
//
#include <hip/hip_runtime.h>

#define N_ROWS 8192
#define DIMS   512
#define MARGIN 0.35f
#define CHUNKS 8

typedef __attribute__((ext_vector_type(16))) __bf16 v16bf;
typedef __attribute__((ext_vector_type(8)))  float  v8f;
typedef __attribute__((ext_vector_type(4)))  int    i4;
typedef __attribute__((ext_vector_type(4)))  float  f4;

union FragBF { i4 q[2]; v16bf bf; };

__device__ __forceinline__ unsigned short f2bf(float f) {
  unsigned u = __float_as_uint(f);
  u += 0x7FFFu + ((u >> 16) & 1u);   // round-to-nearest-even
  return (unsigned short)(u >> 16);
}
__device__ __forceinline__ float bf2f(unsigned short h) {
  return __uint_as_float(((unsigned)h) << 16);
}

// ---------------------------------------------------------------------------
// Kernel 1: per-row stats (sq, cross-entropy) + bf16 hi/lo split of X.
// One wave (32 lanes) per row; 16 elements per lane.
// ---------------------------------------------------------------------------
__global__ __launch_bounds__(256) void k_prep(
    const float* __restrict__ x, const int* __restrict__ lab,
    unsigned short* __restrict__ Ahi, unsigned short* __restrict__ Alo,
    float* __restrict__ sq, float* __restrict__ ce) {
  const int lane = threadIdx.x & 31;
  const int wave = threadIdx.x >> 5;
  const int row  = blockIdx.x * 8 + wave;
  const float* xr = x + (size_t)row * DIMS;

  float v[16];
  {
    const f4* p = (const f4*)(xr + lane * 16);
#pragma unroll
    for (int t = 0; t < 4; ++t) {
      f4 q = p[t];
      v[4 * t + 0] = q.x; v[4 * t + 1] = q.y;
      v[4 * t + 2] = q.z; v[4 * t + 3] = q.w;
    }
  }

  float s = 0.f, mx = -__builtin_inff();
#pragma unroll
  for (int t = 0; t < 16; ++t) { s += v[t] * v[t]; mx = fmaxf(mx, v[t]); }
#pragma unroll
  for (int o = 16; o > 0; o >>= 1) {
    s  += __shfl_xor(s, o, 32);
    mx  = fmaxf(mx, __shfl_xor(mx, o, 32));
  }

  const float inv = 1.0f / sqrtf(s);
  float sqa = 0.f, se = 0.f;
#pragma unroll
  for (int t = 0; t < 16; ++t) {
    float n = v[t] * inv;
    sqa += n * n;
    se  += expf(v[t] - mx);
  }
#pragma unroll
  for (int o = 16; o > 0; o >>= 1) {
    sqa += __shfl_xor(sqa, o, 32);
    se  += __shfl_xor(se, o, 32);
  }
  if (lane == 0) {
    sq[row] = sqa;
    ce[row] = mx + logf(se) - xr[lab[row]];
  }

  // bf16 hi/lo split, 16 elements/lane, packed 128-bit stores.
  union { unsigned short h[8]; i4 q; } h0, h1, l0, l1;
#pragma unroll
  for (int t = 0; t < 8; ++t) {
    unsigned short h = f2bf(v[t]);
    h0.h[t] = h;  l0.h[t] = f2bf(v[t] - bf2f(h));
  }
#pragma unroll
  for (int t = 0; t < 8; ++t) {
    unsigned short h = f2bf(v[8 + t]);
    h1.h[t] = h;  l1.h[t] = f2bf(v[8 + t] - bf2f(h));
  }
  i4* dh = (i4*)(Ahi + (size_t)row * DIMS + lane * 16);
  i4* dl = (i4*)(Alo + (size_t)row * DIMS + lane * 16);
  dh[0] = h0.q; dh[1] = h1.q;
  dl[0] = l0.q; dl[1] = l1.q;
}

// ---------------------------------------------------------------------------
// Kernel 2: fused Gram-tile + masked row max/min via bf16 WMMA (hi/lo split).
// 2x2 register blocking: each wave owns a 32x32 output patch per iteration
// (16 global_load_b128 per 12 v_wmma instead of 32 per 12).
// grid.y splits columns into CHUNKS chunks for wave-level parallelism.
// ---------------------------------------------------------------------------
__global__ __launch_bounds__(256) void k_dist(
    const unsigned short* __restrict__ Ahi, const unsigned short* __restrict__ Alo,
    const float* __restrict__ sq, const int* __restrict__ lab,
    float* __restrict__ posP, float* __restrict__ negP) {
  const int lane  = threadIdx.x & 31;
  const int wave  = threadIdx.x >> 5;
  const int half  = lane >> 4;
  const int nm    = lane & 15;
  const int rbase = blockIdx.x * 256 + wave * 32;   // 2 row strips of 16
  const int chunk = blockIdx.y;
  const int cbeg  = chunk * (N_ROWS / CHUNKS);
  const int cend  = cbeg + (N_ROWS / CHUNKS);

  // A fragment bases (16x32 bf16 A layout: lane-half holds K {h*8..+7} U {h*8+16..+7})
  const unsigned short* aHiB[2];
  const unsigned short* aLoB[2];
#pragma unroll
  for (int r = 0; r < 2; ++r) {
    aHiB[r] = Ahi + (size_t)(rbase + 16 * r + nm) * DIMS + half * 8;
    aLoB[r] = Alo + (size_t)(rbase + 16 * r + nm) * DIMS + half * 8;
  }

  float sqI[2][8]; int labI[2][8];
#pragma unroll
  for (int r = 0; r < 2; ++r)
#pragma unroll
    for (int v = 0; v < 8; ++v) {
      int row = rbase + 16 * r + v + 8 * half;   // C/D layout: VGPR v, lane-half -> M
      sqI[r][v] = sq[row];
      labI[r][v] = lab[row];
    }

  float pos[2][8], neg[2][8];
#pragma unroll
  for (int r = 0; r < 2; ++r)
#pragma unroll
    for (int v = 0; v < 8; ++v) { pos[r][v] = -__builtin_inff(); neg[r][v] = __builtin_inff(); }

  for (int c0 = cbeg; c0 < cend; c0 += 32) {
    const unsigned short* bHiB[2];
    const unsigned short* bLoB[2];
    float sqJ[2]; int labJ[2];
#pragma unroll
    for (int c = 0; c < 2; ++c) {
      const int jcol = c0 + 16 * c + nm;
      // B layout (32x16 bf16): lane-half holds K {h*8..+7} U {h*8+16..+7}
      bHiB[c] = Ahi + (size_t)jcol * DIMS + half * 8;
      bLoB[c] = Alo + (size_t)jcol * DIMS + half * 8;
      sqJ[c]  = sq[jcol];
      labJ[c] = lab[jcol];
    }
    __builtin_prefetch(bHiB[0] + 32 * DIMS, 0, 1);   // next column super-tile
    __builtin_prefetch(bLoB[0] + 32 * DIMS, 0, 1);
    __builtin_prefetch(bHiB[1] + 32 * DIMS, 0, 1);
    __builtin_prefetch(bLoB[1] + 32 * DIMS, 0, 1);

    v8f acc[2][2];
#pragma unroll
    for (int r = 0; r < 2; ++r)
#pragma unroll
      for (int c = 0; c < 2; ++c) acc[r][c] = (v8f){};

#pragma unroll 2
    for (int k0 = 0; k0 < DIMS; k0 += 32) {
      FragBF ahi[2], alo[2], bhi[2], blo[2];
#pragma unroll
      for (int r = 0; r < 2; ++r) {
        ahi[r].q[0] = *(const i4*)(aHiB[r] + k0);  ahi[r].q[1] = *(const i4*)(aHiB[r] + k0 + 16);
        alo[r].q[0] = *(const i4*)(aLoB[r] + k0);  alo[r].q[1] = *(const i4*)(aLoB[r] + k0 + 16);
      }
#pragma unroll
      for (int c = 0; c < 2; ++c) {
        bhi[c].q[0] = *(const i4*)(bHiB[c] + k0);  bhi[c].q[1] = *(const i4*)(bHiB[c] + k0 + 16);
        blo[c].q[0] = *(const i4*)(bLoB[c] + k0);  blo[c].q[1] = *(const i4*)(bLoB[c] + k0 + 16);
      }
      // G ~= hi*hi + hi*lo + lo*hi : 12 WMMAs reuse the 8 fragments above.
#pragma unroll
      for (int r = 0; r < 2; ++r)
#pragma unroll
        for (int c = 0; c < 2; ++c) {
          acc[r][c] = __builtin_amdgcn_wmma_f32_16x16x32_bf16(false, ahi[r].bf, false, bhi[c].bf, (short)0, acc[r][c], false, false);
          acc[r][c] = __builtin_amdgcn_wmma_f32_16x16x32_bf16(false, ahi[r].bf, false, blo[c].bf, (short)0, acc[r][c], false, false);
          acc[r][c] = __builtin_amdgcn_wmma_f32_16x16x32_bf16(false, alo[r].bf, false, bhi[c].bf, (short)0, acc[r][c], false, false);
        }
    }

#pragma unroll
    for (int r = 0; r < 2; ++r)
#pragma unroll
      for (int c = 0; c < 2; ++c)
#pragma unroll
        for (int v = 0; v < 8; ++v) {
          float d = sqI[r][v] + sqJ[c] - 2.0f * acc[r][c][v];
          if (labI[r][v] == labJ[c]) pos[r][v] = fmaxf(pos[r][v], d);
          else                       neg[r][v] = fminf(neg[r][v], d);
        }
  }

  // Reduce across the 16 lanes (N dimension) within each lane-half.
#pragma unroll
  for (int o = 1; o < 16; o <<= 1) {
#pragma unroll
    for (int r = 0; r < 2; ++r)
#pragma unroll
      for (int v = 0; v < 8; ++v) {
        pos[r][v] = fmaxf(pos[r][v], __shfl_xor(pos[r][v], o, 32));
        neg[r][v] = fminf(neg[r][v], __shfl_xor(neg[r][v], o, 32));
      }
  }
  if (nm == 0) {
#pragma unroll
    for (int r = 0; r < 2; ++r)
#pragma unroll
      for (int v = 0; v < 8; ++v) {
        int row = rbase + 16 * r + v + 8 * half;
        posP[row * CHUNKS + chunk] = pos[r][v];
        negP[row * CHUNKS + chunk] = neg[r][v];
      }
  }
}

// ---------------------------------------------------------------------------
// Kernel 3: deterministic final reduction -> scalar loss.
// ---------------------------------------------------------------------------
__global__ __launch_bounds__(256) void k_final(
    const float* __restrict__ posP, const float* __restrict__ negP,
    const float* __restrict__ ce, float* __restrict__ out) {
  __shared__ float sh[256];
  float acc = 0.f;
  for (int r = threadIdx.x; r < N_ROWS; r += 256) {
    float p = -__builtin_inff(), n = __builtin_inff();
#pragma unroll
    for (int c = 0; c < CHUNKS; ++c) {
      p = fmaxf(p, posP[r * CHUNKS + c]);
      n = fminf(n, negP[r * CHUNKS + c]);
    }
    acc += fmaxf(p - n + MARGIN, 0.f) + ce[r];
  }
  sh[threadIdx.x] = acc;
  __syncthreads();
  for (int o = 128; o > 0; o >>= 1) {
    if ((int)threadIdx.x < o) sh[threadIdx.x] += sh[threadIdx.x + o];
    __syncthreads();
  }
  if (threadIdx.x == 0) out[0] = sh[0] / (float)N_ROWS;
}

// ---------------------------------------------------------------------------
extern "C" void kernel_launch(void* const* d_in, const int* in_sizes, int n_in,
                              void* d_out, int out_size, void* d_ws, size_t ws_size,
                              hipStream_t stream) {
  (void)in_sizes; (void)n_in; (void)out_size; (void)ws_size;
  const float* x   = (const float*)d_in[0];
  const int*   lab = (const int*)d_in[1];
  float*       out = (float*)d_out;

  char* ws = (char*)d_ws;
  const size_t BF = (size_t)N_ROWS * DIMS * sizeof(unsigned short);  // 8 MB
  unsigned short* Ahi  = (unsigned short*)(ws);
  unsigned short* Alo  = (unsigned short*)(ws + BF);
  float*          sq   = (float*)(ws + 2 * BF);
  float*          ce   = (float*)(ws + 2 * BF + (size_t)N_ROWS * 4);
  float*          posP = (float*)(ws + 2 * BF + 2 * (size_t)N_ROWS * 4);
  float*          negP = (float*)(ws + 2 * BF + 2 * (size_t)N_ROWS * 4 + (size_t)CHUNKS * N_ROWS * 4);

  k_prep<<<N_ROWS / 8, 256, 0, stream>>>(x, lab, Ahi, Alo, sq, ce);
  dim3 grid(N_ROWS / 256, CHUNKS);
  k_dist<<<grid, 256, 0, stream>>>(Ahi, Alo, sq, lab, posP, negP);
  k_final<<<1, 256, 0, stream>>>(posP, negP, ce, out);
}
